// Model_9844065042802
// MI455X (gfx1250) — matrix operand order repaired and checked by reading, not compiled, not run on hardware
//
#include <hip/hip_runtime.h>
#include <hip/hip_bf16.h>

typedef __attribute__((ext_vector_type(16))) _Float16 v16h;
typedef __attribute__((ext_vector_type(8)))  _Float16 v8h;
typedef __attribute__((ext_vector_type(8)))  float    v8f;

#define DD 64
#define NUU 100000
#define NII 50000
#define EE 500000
#define NQ 8192
#define NLAYER 3
#define SLOPE 0.2f
#define EPSF 1e-12f

#define AS_STRIDE 72   // f16 LDS row stride (144 B, 16B-aligned rows, pads banks)
#define PS_STRIDE 68   // f32 LDS row stride

// ---------------------------------------------------------------------------
// CDNA5 async global->LDS copy (16 B per lane), tracked by ASYNCcnt.
// ---------------------------------------------------------------------------
__device__ __forceinline__ void async_copy_b128(void* lds_dst, const void* gsrc) {
    unsigned loff = (unsigned)(uintptr_t)lds_dst;   // low 32 bits == LDS offset
    asm volatile("global_load_async_to_lds_b128 %0, %1, off"
                 :: "v"(loff), "v"(gsrc) : "memory");
}
__device__ __forceinline__ void async_wait0() {
    asm volatile("s_wait_asynccnt 0x0" ::: "memory");
}

// Build A fragment (16 halves) from a row-major f16 LDS row per ISA 7.12.2.
__device__ __forceinline__ v16h a_frag(const _Float16* arow, int kk, int hih) {
    v8h lo = *(const v8h*)(arow + kk + hih * 8);
    v8h hi = *(const v8h*)(arow + kk + 16 + hih * 8);
    return __builtin_shufflevector(lo, hi, 0,1,2,3,4,5,6,7,8,9,10,11,12,13,14,15);
}

// ---------------------------------------------------------------------------
// Setup kernels
// ---------------------------------------------------------------------------
__global__ __launch_bounds__(256) void zero_kernel(float* p, int n) {
    int i = blockIdx.x * 256 + threadIdx.x;
    if (i < n) p[i] = 0.f;
}

__global__ __launch_bounds__(256) void degree_kernel(const int* __restrict__ u_idx,
                                                     const int* __restrict__ i_idx,
                                                     float* deg_u, float* deg_i) {
    int e = blockIdx.x * 256 + threadIdx.x;
    if (e < EE) {
        atomicAdd(&deg_u[u_idx[e]], 1.f);
        atomicAdd(&deg_i[i_idx[e]], 1.f);
    }
}

__global__ __launch_bounds__(256) void norm_kernel(const int* __restrict__ u_idx,
                                                   const int* __restrict__ i_idx,
                                                   const float* __restrict__ deg_u,
                                                   const float* __restrict__ deg_i,
                                                   float* __restrict__ nrm) {
    int e = blockIdx.x * 256 + threadIdx.x;
    if (e < EE) nrm[e] = rsqrtf(deg_u[u_idx[e]] * deg_i[i_idx[e]]);
}

__global__ __launch_bounds__(256) void cvt_f16_kernel(const float* __restrict__ src,
                                                      _Float16* __restrict__ dst, int n) {
    int i = blockIdx.x * 256 + threadIdx.x;
    if (i < n) dst[i] = (_Float16)src[i];
}

// Pre-swizzle W1[l], W2[l] (f32 row-major KxN) into the exact WMMA B-operand
// layout as f16: Wsw[chunk=K/32][n/16][lane=((K>>3)&1)*16|(n&15)][j=((K>>4)&1)*8|(K&7)].
__global__ __launch_bounds__(256) void swizzleW_kernel(const float* __restrict__ W1,
                                                       const float* __restrict__ W2,
                                                       _Float16* __restrict__ Wsw) {
    int id = blockIdx.x * 256 + threadIdx.x;        // 6 * 4096 elements
    if (id >= 6 * 4096) return;
    int m = id >> 12;                               // matrix index: 2*l + which
    int e = id & 4095;
    int l = m >> 1, which = m & 1;
    const float* src = (which ? W2 : W1) + l * 4096;
    int k = e >> 6, c = e & 63;
    int off = (((k >> 5) * 4 + (c >> 4)) * 32 + (((k >> 3) & 1) * 16 + (c & 15))) * 16
              + (((k >> 4) & 1) * 8 + (k & 7));
    Wsw[m * 4096 + off] = (_Float16)src[e];
}

// ---------------------------------------------------------------------------
// Node transform: P = H @ W + b (also seeds Acc = P).  64x64 tile per block,
// 8 wave32s, 2 WMMA tiles per wave, A/B staged via async-to-LDS.
// ---------------------------------------------------------------------------
__global__ __launch_bounds__(256)
void node_gemm_kernel(const _Float16* __restrict__ Hh, const _Float16* __restrict__ Wsw,
                      const float* __restrict__ bias,
                      float* __restrict__ P, float* __restrict__ Acc, int nRows) {
    __shared__ __align__(16) _Float16 As[DD][AS_STRIDE];
    __shared__ __align__(32) _Float16 Bsw[2][4][32][16];   // 8 KB, B-operand order
    __shared__ float bsm[DD];

    const int tid = threadIdx.x;
    const int rowBase = blockIdx.x * DD;

    if (tid < DD) bsm[tid] = bias[tid];
#pragma unroll
    for (int i = 0; i < 2; ++i) {                  // A rows: 64 x 128 B
        int idx = tid + i * 256;                   // 0..511 segments of 16 B
        int r = idx >> 3, seg = idx & 7;
        int gr = rowBase + r; if (gr >= nRows) gr = nRows - 1;
        async_copy_b128(&As[r][seg * 8], Hh + gr * DD + seg * 8);
    }
#pragma unroll
    for (int i = 0; i < 2; ++i) {                  // B: 8 KB straight copy
        int idx = tid + i * 256;
        async_copy_b128((char*)Bsw + idx * 16, (const char*)Wsw + idx * 16);
    }
    async_wait0();
    __syncthreads();

    const int lane = tid & 31;
    const int wave = tid >> 5;
    const int hih  = lane >> 4;
    const int ln   = lane & 15;

#pragma unroll
    for (int t = 0; t < 2; ++t) {
        const int tile = wave + t * 8;
        const int tm = (tile >> 2) * 16;
        const int tn = (tile & 3) * 16;
        const _Float16* arow = &As[tm + ln][0];
        v8f acc = {};
#pragma unroll
        for (int c = 0; c < 2; ++c) {
            v16h a = a_frag(arow, c * 32, hih);
            v16h b = *(const v16h*)&Bsw[c][tn >> 4][lane][0];
            acc = __builtin_amdgcn_wmma_f32_16x16x32_f16(false, a, false, b,
                                                         (short)0, acc, false, false);
        }
        const float bn = bsm[tn + ln];
        if (rowBase + DD <= nRows) {               // uniform fast path
#pragma unroll
            for (int r = 0; r < 8; ++r) {
                const int o = (rowBase + tm + hih * 8 + r) * DD + tn + ln;  // 32-bit
                float val = acc[r] + bn;
                P[o] = val;  Acc[o] = val;
            }
        } else {
#pragma unroll
            for (int r = 0; r < 8; ++r) {
                const int gr = rowBase + tm + hih * 8 + r;
                if (gr < nRows) {
                    const int o = gr * DD + tn + ln;
                    float val = acc[r] + bn;
                    P[o] = val;  Acc[o] = val;
                }
            }
        }
    }
}

// ---------------------------------------------------------------------------
// Fused edge kernel: async-gather f16 rows of hu[u], hi[i] -> A = eu*ei via
// v_pk_mul_f16 during operand build -> WMMA @ W2 -> normalized message
// scatter-add into acc_u/acc_i with f32 global atomics.  64 edges per block.
// ---------------------------------------------------------------------------
__global__ __launch_bounds__(256)
void edge_kernel(const _Float16* __restrict__ huh, const _Float16* __restrict__ hih16,
                 const float* __restrict__ pu, const float* __restrict__ pi,
                 const int* __restrict__ u_idx, const int* __restrict__ i_idx,
                 const float* __restrict__ nrm,
                 const _Float16* __restrict__ Wsw2, const float* __restrict__ b2,
                 float* __restrict__ acc_u, float* __restrict__ acc_i) {
    __shared__ __align__(16) _Float16 EU[DD][AS_STRIDE];
    __shared__ __align__(16) _Float16 EI[DD][AS_STRIDE];
    __shared__ __align__(32) _Float16 Bsw[2][4][32][16];
    __shared__ float Ps[DD][PS_STRIDE];
    __shared__ float bsm[DD];
    __shared__ float sn[DD];
    __shared__ int   su[DD], si[DD];

    const int tid = threadIdx.x;
    const int eBase = blockIdx.x * DD;

    if (tid < DD) {
        const int e = eBase + tid;
        bsm[tid] = b2[tid];
        if (e < EE) { su[tid] = u_idx[e]; si[tid] = i_idx[e]; sn[tid] = nrm[e]; }
        else        { su[tid] = 0;        si[tid] = 0;        sn[tid] = 0.f;   }
    }
    __syncthreads();

#pragma unroll
    for (int i = 0; i < 2; ++i) {                  // gather eu / ei rows (f16)
        int idx = tid + i * 256;                   // 0..511, 16 B each
        int r = idx >> 3, seg = idx & 7;
        async_copy_b128(&EU[r][seg * 8], huh   + su[r] * DD + seg * 8);
        async_copy_b128(&EI[r][seg * 8], hih16 + si[r] * DD + seg * 8);
    }
#pragma unroll
    for (int i = 0; i < 2; ++i) {                  // W2 swizzled, 8 KB
        int idx = tid + i * 256;
        async_copy_b128((char*)Bsw + idx * 16, (const char*)Wsw2 + idx * 16);
    }
    async_wait0();
    __syncthreads();

    const int lane = tid & 31;
    const int wave = tid >> 5;
    const int hih  = lane >> 4;
    const int ln   = lane & 15;

#pragma unroll
    for (int t = 0; t < 2; ++t) {
        const int tile = wave + t * 8;
        const int tm = (tile >> 2) * 16;
        const int tn = (tile & 3) * 16;
        const _Float16* er = &EU[tm + ln][0];
        const _Float16* fr = &EI[tm + ln][0];
        v8f acc = {};
#pragma unroll
        for (int c = 0; c < 2; ++c) {
            const int kk = c * 32;
            v8h lo = *(const v8h*)(er + kk + hih * 8) * *(const v8h*)(fr + kk + hih * 8);
            v8h hi = *(const v8h*)(er + kk + 16 + hih * 8) *
                     *(const v8h*)(fr + kk + 16 + hih * 8);
            v16h a = __builtin_shufflevector(lo, hi,
                        0,1,2,3,4,5,6,7,8,9,10,11,12,13,14,15);
            v16h b = *(const v16h*)&Bsw[c][tn >> 4][lane][0];
            acc = __builtin_amdgcn_wmma_f32_16x16x32_f16(false, a, false, b,
                                                         (short)0, acc, false, false);
        }
#pragma unroll
        for (int r = 0; r < 8; ++r)
            Ps[tm + hih * 8 + r][tn + ln] = acc[r];
    }
    __syncthreads();

    // Scatter normalized messages (L2-resident f32 atomics, 32-bit offsets)
    if (eBase + DD <= EE) {
#pragma unroll
        for (int i = 0; i < 16; ++i) {
            int idx = tid + i * 256;
            int r = idx >> 6, c = idx & 63;
            const float pw = Ps[r][c] + bsm[c];
            const int u = su[r], it = si[r];
            const float nr = sn[r];
            atomicAdd(&acc_i[it * DD + c], nr * (pu[u * DD + c] + pw));
            atomicAdd(&acc_u[u * DD + c],  nr * (pi[it * DD + c] + pw));
        }
    } else {
#pragma unroll
        for (int i = 0; i < 16; ++i) {
            int idx = tid + i * 256;
            int r = idx >> 6, c = idx & 63;
            if (eBase + r < EE) {
                const float pw = Ps[r][c] + bsm[c];
                const int u = su[r], it = si[r];
                const float nr = sn[r];
                atomicAdd(&acc_i[it * DD + c], nr * (pu[u * DD + c] + pw));
                atomicAdd(&acc_u[u * DD + c],  nr * (pi[it * DD + c] + pw));
            }
        }
    }
}

// ---------------------------------------------------------------------------
// Post: h = l2norm(leaky_relu(acc)); writes f32 + f16 mirror.  Wave32 per row.
// ---------------------------------------------------------------------------
__global__ __launch_bounds__(256)
void post_kernel(const float* __restrict__ acc, float* __restrict__ h,
                 _Float16* __restrict__ hh, int nRows) {
    const int tid  = threadIdx.x;
    const int lane = tid & 31;
    const int row  = blockIdx.x * 8 + (tid >> 5);
    if (row >= nRows) return;
    const int o0 = row * DD + lane;
    const int o1 = o0 + 32;
    float v0 = acc[o0];
    float v1 = acc[o1];
    v0 = v0 > 0.f ? v0 : SLOPE * v0;
    v1 = v1 > 0.f ? v1 : SLOPE * v1;
    float s = v0 * v0 + v1 * v1;
#pragma unroll
    for (int o = 16; o > 0; o >>= 1) s += __shfl_xor(s, o, 32);
    const float inv = 1.f / fmaxf(sqrtf(s), EPSF);
    const float r0 = v0 * inv, r1 = v1 * inv;
    h[o0]  = r0;
    h[o1]  = r1;
    hh[o0] = (_Float16)r0;
    hh[o1] = (_Float16)r1;
}

// ---------------------------------------------------------------------------
// Gather one 64-col layer slice of selected rows into the 256-col output
// ---------------------------------------------------------------------------
__global__ __launch_bounds__(256)
void gather_kernel(const float* __restrict__ src, const int* __restrict__ idxArr,
                   float* __restrict__ outBase, int colOff) {
    const int tid = threadIdx.x;
    const int j  = blockIdx.x * 16 + (tid >> 4);
    const int c4 = tid & 15;
    if (j < NQ) {
        float4 v = ((const float4*)(src + idxArr[j] * DD))[c4];
        ((float4*)(outBase + j * 256 + colOff))[c4] = v;
    }
}

// ---------------------------------------------------------------------------
extern "C" void kernel_launch(void* const* d_in, const int* in_sizes, int n_in,
                              void* d_out, int out_size, void* d_ws, size_t ws_size,
                              hipStream_t stream) {
    const float* feat_user = (const float*)d_in[0];
    const float* feat_item = (const float*)d_in[1];
    const float* W1        = (const float*)d_in[2];
    const float* b1        = (const float*)d_in[3];
    const float* W2        = (const float*)d_in[4];
    const float* b2        = (const float*)d_in[5];
    const int*   u_idx     = (const int*)d_in[6];
    const int*   i_idx     = (const int*)d_in[7];
    const int*   users     = (const int*)d_in[8];
    const int*   pos_items = (const int*)d_in[9];
    const int*   neg_items = (const int*)d_in[10];

    // workspace carve-up
    float* ws    = (float*)d_ws;
    float* deg_u = ws;                 ws += NUU;
    float* deg_i = ws;                 ws += NII;
    float* nrm   = ws;                 ws += EE;
    float* hu    = ws;                 ws += (size_t)NUU * DD;
    float* hiv   = ws;                 ws += (size_t)NII * DD;
    float* pu    = ws;                 ws += (size_t)NUU * DD;
    float* pi    = ws;                 ws += (size_t)NII * DD;
    float* acc_u = ws;                 ws += (size_t)NUU * DD;
    float* acc_i = ws;                 ws += (size_t)NII * DD;
    _Float16* hws   = (_Float16*)ws;                  // 16B-aligned by construction
    _Float16* Wsw   = hws;             hws += 6 * 4096;
    _Float16* huh   = hws;             hws += (size_t)NUU * DD;
    _Float16* hih16 = hws;             hws += (size_t)NII * DD;

    float* out_user = (float*)d_out;
    float* out_pos  = out_user + (size_t)NQ * 256;
    float* out_neg  = out_pos  + (size_t)NQ * 256;

    const int eBlocks  = (EE + 255) / 256;
    const int ubBlocks = (NUU + DD - 1) / DD;   // 1563
    const int ibBlocks = (NII + DD - 1) / DD;   // 782
    const int edBlocks = (EE + DD - 1) / DD;    // 7813
    const int puBlocks = (NUU + 7) / 8;
    const int piBlocks = (NII + 7) / 8;
    const int gBlocks  = NQ / 16;

    // degrees + norm + swizzled weights
    zero_kernel<<<(NUU + NII + 255) / 256, 256, 0, stream>>>(deg_u, NUU + NII);
    degree_kernel<<<eBlocks, 256, 0, stream>>>(u_idx, i_idx, deg_u, deg_i);
    norm_kernel<<<eBlocks, 256, 0, stream>>>(u_idx, i_idx, deg_u, deg_i, nrm);
    swizzleW_kernel<<<(6 * 4096 + 255) / 256, 256, 0, stream>>>(W1, W2, Wsw);

    // h0 = features (f32 working copy + f16 mirror)
    hipMemcpyAsync(hu,  feat_user, (size_t)NUU * DD * sizeof(float),
                   hipMemcpyDeviceToDevice, stream);
    hipMemcpyAsync(hiv, feat_item, (size_t)NII * DD * sizeof(float),
                   hipMemcpyDeviceToDevice, stream);
    cvt_f16_kernel<<<(NUU * DD + 255) / 256, 256, 0, stream>>>(feat_user, huh, NUU * DD);
    cvt_f16_kernel<<<(NII * DD + 255) / 256, 256, 0, stream>>>(feat_item, hih16, NII * DD);

    // layer-0 slice of the concatenated output
    gather_kernel<<<gBlocks, 256, 0, stream>>>(hu,  users,     out_user, 0);
    gather_kernel<<<gBlocks, 256, 0, stream>>>(hiv, pos_items, out_pos,  0);
    gather_kernel<<<gBlocks, 256, 0, stream>>>(hiv, neg_items, out_neg,  0);

    for (int l = 0; l < NLAYER; ++l) {
        const _Float16* w1sw = Wsw + (2 * l) * 4096;
        const _Float16* w2sw = Wsw + (2 * l + 1) * 4096;
        const float* bb1 = b1 + l * DD;
        const float* bb2 = b2 + l * DD;

        node_gemm_kernel<<<ubBlocks, 256, 0, stream>>>(huh,   w1sw, bb1, pu, acc_u, NUU);
        node_gemm_kernel<<<ibBlocks, 256, 0, stream>>>(hih16, w1sw, bb1, pi, acc_i, NII);

        edge_kernel<<<edBlocks, 256, 0, stream>>>(huh, hih16, pu, pi, u_idx, i_idx, nrm,
                                                  w2sw, bb2, acc_u, acc_i);

        post_kernel<<<puBlocks, 256, 0, stream>>>(acc_u, hu,  huh,   NUU);
        post_kernel<<<piBlocks, 256, 0, stream>>>(acc_i, hiv, hih16, NII);

        gather_kernel<<<gBlocks, 256, 0, stream>>>(hu,  users,     out_user, (l + 1) * DD);
        gather_kernel<<<gBlocks, 256, 0, stream>>>(hiv, pos_items, out_pos,  (l + 1) * DD);
        gather_kernel<<<gBlocks, 256, 0, stream>>>(hiv, neg_items, out_neg,  (l + 1) * DD);
    }
}
